// MultimodalMultiheadAttention_64682207478545
// MI455X (gfx1250) — matrix operand order repaired and checked by reading, not compiled, not run on hardware
//
#include <hip/hip_runtime.h>
#include <stdint.h>

#define AS_GLOBAL __attribute__((address_space(1)))
#define AS_LOCAL  __attribute__((address_space(3)))

typedef __attribute__((ext_vector_type(4)))  int    v4i;

// -------- CDNA5 async global->LDS path detection ---------------------------
#if defined(__has_builtin)
#  if __has_builtin(__builtin_amdgcn_global_load_async_to_lds_b128)
#    define USE_ASYNC_LDS 1
#  endif
#endif
#ifndef USE_ASYNC_LDS
#  define USE_ASYNC_LDS 0
#  warning "gfx1250 async-to-LDS builtin not available; falling back to register staging"
#endif

__device__ __forceinline__ void wait_async_all() {
#if USE_ASYNC_LDS
#  if __has_builtin(__builtin_amdgcn_s_wait_asynccnt)
  __builtin_amdgcn_s_wait_asynccnt(0);
#  else
  asm volatile("s_wait_asynccnt 0x0" ::: "memory");
#  endif
#endif
}

#if USE_ASYNC_LDS
__device__ __forceinline__ void async_b128(const unsigned short* g, unsigned short* l) {
  // GLOBAL_LOAD_ASYNC_TO_LDS_B128: per-lane 16B global -> LDS, tracked by ASYNCcnt
  __builtin_amdgcn_global_load_async_to_lds_b128(
      (AS_GLOBAL v4i*)g, (AS_LOCAL v4i*)l, 0, 0);
}
#endif

// ---------------------------------------------------------------------------
// Types for CDNA5 WMMA (wave32): 16x16x32 bf16 -> f32 accumulate
// ---------------------------------------------------------------------------
typedef __attribute__((ext_vector_type(16))) __bf16 v16bf;
typedef __attribute__((ext_vector_type(8)))  float  v8f;

__device__ __forceinline__ unsigned short f32_to_bf16(float f) {
  unsigned int u = __float_as_uint(f);
  u += 0x7fffu + ((u >> 16) & 1u);          // round-to-nearest-even
  return (unsigned short)(u >> 16);
}
__device__ __forceinline__ void unpack8(uint4 u, float* f) {
  f[0] = __uint_as_float(u.x << 16); f[1] = __uint_as_float(u.x & 0xffff0000u);
  f[2] = __uint_as_float(u.y << 16); f[3] = __uint_as_float(u.y & 0xffff0000u);
  f[4] = __uint_as_float(u.z << 16); f[5] = __uint_as_float(u.z & 0xffff0000u);
  f[6] = __uint_as_float(u.w << 16); f[7] = __uint_as_float(u.w & 0xffff0000u);
}
__device__ __forceinline__ void store_val(float* p, float v)          { *p = v; }
__device__ __forceinline__ void store_val(unsigned short* p, float v) { *p = f32_to_bf16(v); }

// ---------------------------------------------------------------------------
// Elementwise f32 -> bf16 conversion (vectorized by 4)
// ---------------------------------------------------------------------------
__global__ __launch_bounds__(256) void cvt_f32_to_bf16(
    const float* __restrict__ in, unsigned short* __restrict__ out, int n4) {
  int i = blockIdx.x * 256 + threadIdx.x;
  if (i >= n4) return;
  float4 f = reinterpret_cast<const float4*>(in)[i];
  uint2 o;
  o.x = (unsigned)f32_to_bf16(f.x) | ((unsigned)f32_to_bf16(f.y) << 16);
  o.y = (unsigned)f32_to_bf16(f.z) | ((unsigned)f32_to_bf16(f.w) << 16);
  reinterpret_cast<uint2*>(out)[i] = o;
}

// ---------------------------------------------------------------------------
// Generic GEMM: C[M,N] = A[M,K] * B[N,K]^T + bias[N]
// bf16 operands, f32 accumulate via v_wmma_f32_16x16x32_bf16.
// Block tile 128x256x32, 256 threads = 8 waves arranged 2x4; each wave owns a
// 64x64 sub-tile = 4x4 WMMA tiles (16 WMMA per k-step, 16 ds_load_b128).
// LDS double-buffered; staging via async global->LDS when available.
// Requires M%128==0, N%256==0, K%32==0.
// ---------------------------------------------------------------------------
#define BM 128
#define BN 256
#define BK 32
#define LDK 40   // padded LDS row stride (bf16 elems): 80 B, 16 B aligned, conflict-free

template <typename OutT>
__global__ __launch_bounds__(256) void gemm_bf16_wmma(
    const unsigned short* __restrict__ A,   // M x K (bf16 bits)
    const unsigned short* __restrict__ Bm,  // N x K (bf16 bits)
    const float* __restrict__ bias,         // N
    OutT* __restrict__ C,                   // M x N
    int N, int K) {
  __shared__ unsigned short As[2][BM * LDK];   // 2 x 10 KiB
  __shared__ unsigned short Bs[2][BN * LDK];   // 2 x 20 KiB

  const int tid  = threadIdx.x;
  const int lane = tid & 31;
  const int wave = tid >> 5;
  const int lrow = lane & 15;
  const int l16  = lane >> 4;
  const int wm   = wave >> 2;   // 0..1 : 64-row group
  const int wn   = wave & 3;    // 0..3 : 64-col group

  const int blockM = blockIdx.y * BM;
  const int blockN = blockIdx.x * BN;

  // staging: A -> each thread moves one 16-elem half-row (2 x b128)
  //          B -> each thread moves one full 32-elem row  (4 x b128)
  const int aRow  = tid >> 1;
  const int aHalf = tid & 1;
  const size_t aOff = (size_t)(blockM + aRow) * K + aHalf * 16;
  const size_t bOff = (size_t)(blockN + tid) * K;
  const int aLds = aRow * LDK + aHalf * 16;
  const int bLds = tid * LDK;

#if !USE_ASYNC_LDS
  uint4 ra[2], rb[4];
#endif

  auto stage = [&](int kcol, int b) {
#if USE_ASYNC_LDS
    unsigned short* Asb = &As[b][0];
    unsigned short* Bsb = &Bs[b][0];
    async_b128(A + aOff + kcol,      &Asb[aLds]);
    async_b128(A + aOff + kcol + 8,  &Asb[aLds + 8]);
    async_b128(Bm + bOff + kcol,      &Bsb[bLds]);
    async_b128(Bm + bOff + kcol + 8,  &Bsb[bLds + 8]);
    async_b128(Bm + bOff + kcol + 16, &Bsb[bLds + 16]);
    async_b128(Bm + bOff + kcol + 24, &Bsb[bLds + 24]);
#else
    (void)b;
    ra[0] = *reinterpret_cast<const uint4*>(A + aOff + kcol);
    ra[1] = *reinterpret_cast<const uint4*>(A + aOff + kcol + 8);
    rb[0] = *reinterpret_cast<const uint4*>(Bm + bOff + kcol);
    rb[1] = *reinterpret_cast<const uint4*>(Bm + bOff + kcol + 8);
    rb[2] = *reinterpret_cast<const uint4*>(Bm + bOff + kcol + 16);
    rb[3] = *reinterpret_cast<const uint4*>(Bm + bOff + kcol + 24);
#endif
  };
#if !USE_ASYNC_LDS
  auto commit = [&](int b) {
    unsigned short* Asb = &As[b][0];
    unsigned short* Bsb = &Bs[b][0];
    *reinterpret_cast<uint4*>(&Asb[aLds])     = ra[0];
    *reinterpret_cast<uint4*>(&Asb[aLds + 8]) = ra[1];
    *reinterpret_cast<uint4*>(&Bsb[bLds])      = rb[0];
    *reinterpret_cast<uint4*>(&Bsb[bLds + 8])  = rb[1];
    *reinterpret_cast<uint4*>(&Bsb[bLds + 16]) = rb[2];
    *reinterpret_cast<uint4*>(&Bsb[bLds + 24]) = rb[3];
  };
#endif

  v8f acc[4][4];
  v8f zero = {};
#pragma unroll
  for (int i = 0; i < 4; ++i)
#pragma unroll
    for (int j = 0; j < 4; ++j) acc[i][j] = zero;

  // stage k-tile 0 into buffer 0
  stage(0, 0);
#if !USE_ASYNC_LDS
  commit(0);
#endif

  int buf = 0;
  for (int kt = 0; kt < K; kt += BK) {
    wait_async_all();           // this wave's async copies into LDS complete
    __syncthreads();            // all waves' copies visible; prev reads done

    const int nxt = kt + BK;
    if (nxt < K) stage(nxt, buf ^ 1);   // overlap next tile with compute

    const unsigned short* Asb = &As[buf][0];
    const unsigned short* Bsb = &Bs[buf][0];

    // Fragment layouts per ISA 7.12.2:
    //  A (16-bit, 16x32): lane<16 K {0..7,16..23}; lane>=16 K {8..15,24..31}; M=lane%16
    //  B (16-bit, 32x16): lane<16 K {0..7,16..23}; lane>=16 K {8..15,24..31}; N=lane%16
    union Frag { uint4 q[2]; v16bf v; };
    Frag af[4], bfr[4];
#pragma unroll
    for (int i = 0; i < 4; ++i) {
      int row  = wm * 64 + i * 16 + lrow;
      int base = row * LDK + l16 * 8;
      af[i].q[0] = *reinterpret_cast<const uint4*>(&Asb[base]);
      af[i].q[1] = *reinterpret_cast<const uint4*>(&Asb[base + 16]);
    }
#pragma unroll
    for (int j = 0; j < 4; ++j) {
      int col  = wn * 64 + j * 16 + lrow;
      int base = col * LDK + l16 * 8;
      bfr[j].q[0] = *reinterpret_cast<const uint4*>(&Bsb[base]);
      bfr[j].q[1] = *reinterpret_cast<const uint4*>(&Bsb[base + 16]);
    }

#pragma unroll
    for (int i = 0; i < 4; ++i)
#pragma unroll
      for (int j = 0; j < 4; ++j)
        acc[i][j] = __builtin_amdgcn_wmma_f32_16x16x32_bf16(
            false, af[i].v, false, bfr[j].v, (short)0, acc[i][j], false, false);

#if !USE_ASYNC_LDS
    if (nxt < K) commit(buf ^ 1);
#endif
    buf ^= 1;
  }

  // epilogue: C/D layout -> VGPR r maps to row r + 8*l16, col = lane%16
#pragma unroll
  for (int j = 0; j < 4; ++j) {
    int col  = blockN + wn * 64 + j * 16 + lrow;
    float bv = bias[col];
#pragma unroll
    for (int i = 0; i < 4; ++i) {
      int row0 = blockM + wm * 64 + i * 16 + l16 * 8;
#pragma unroll
      for (int r = 0; r < 8; ++r) {
        float v = acc[i][j][r] + bv;
        store_val(&C[(size_t)(row0 + r) * N + col], v);
      }
    }
  }
}

// ---------------------------------------------------------------------------
// Tiny 3x3 cross-modal attention per (token, head). One thread per (p,h):
// reads q,k,v (3 modalities x 64, bf16), f32 softmax, writes vals row (bf16)
// in the reference's [h][m][d] packing. 262144 threads total.
// ---------------------------------------------------------------------------
__global__ __launch_bounds__(256) void attn3x3_kernel(
    const unsigned short* __restrict__ qkv0,
    const unsigned short* __restrict__ qkv1,
    const unsigned short* __restrict__ qkv2,
    unsigned short* __restrict__ vals) {
  const int gid = blockIdx.x * 256 + threadIdx.x;
  const int h   = gid & 15;
  const int p   = gid >> 4;
  const size_t rowOff = (size_t)p * 3072 + h * 192;  // per-head packed q|k|v
  const unsigned short* base[3] = {qkv0 + rowOff, qkv1 + rowOff, qkv2 + rowOff};

  float dot[3][3];
#pragma unroll
  for (int m = 0; m < 3; ++m)
#pragma unroll
    for (int n = 0; n < 3; ++n) dot[m][n] = 0.f;

#pragma unroll
  for (int c = 0; c < 8; ++c) {
    float kf[3][8];
#pragma unroll
    for (int n = 0; n < 3; ++n)
      unpack8(*reinterpret_cast<const uint4*>(base[n] + 64 + c * 8), kf[n]);
#pragma unroll
    for (int m = 0; m < 3; ++m) {
      float qf[8];
      unpack8(*reinterpret_cast<const uint4*>(base[m] + c * 8), qf);
#pragma unroll
      for (int n = 0; n < 3; ++n) {
        float s = 0.f;
#pragma unroll
        for (int d = 0; d < 8; ++d) s += qf[d] * kf[n][d];
        dot[m][n] += s;
      }
    }
  }

  float a[3][3];
#pragma unroll
  for (int m = 0; m < 3; ++m) {
    float l0 = dot[m][0] * 0.125f, l1 = dot[m][1] * 0.125f, l2 = dot[m][2] * 0.125f;
    float mx = fmaxf(l0, fmaxf(l1, l2));
    float e0 = __expf(l0 - mx), e1 = __expf(l1 - mx), e2 = __expf(l2 - mx);
    float inv = 1.f / (e0 + e1 + e2);
    a[m][0] = e0 * inv; a[m][1] = e1 * inv; a[m][2] = e2 * inv;
  }

  unsigned short* orow = vals + (size_t)p * 3072 + h * 192;  // [h][m][d]
#pragma unroll
  for (int c = 0; c < 8; ++c) {
    float vf[3][8];
#pragma unroll
    for (int n = 0; n < 3; ++n)
      unpack8(*reinterpret_cast<const uint4*>(base[n] + 128 + c * 8), vf[n]);
#pragma unroll
    for (int m = 0; m < 3; ++m) {
      unsigned int w[4];
#pragma unroll
      for (int dd = 0; dd < 4; ++dd) {
        float v0 = a[m][0] * vf[0][2 * dd]     + a[m][1] * vf[1][2 * dd]     +
                   a[m][2] * vf[2][2 * dd]     + vf[m][2 * dd];       // residual +V
        float v1 = a[m][0] * vf[0][2 * dd + 1] + a[m][1] * vf[1][2 * dd + 1] +
                   a[m][2] * vf[2][2 * dd + 1] + vf[m][2 * dd + 1];
        w[dd] = (unsigned)f32_to_bf16(v0) | ((unsigned)f32_to_bf16(v1) << 16);
      }
      uint4 o; o.x = w[0]; o.y = w[1]; o.z = w[2]; o.w = w[3];
      *reinterpret_cast<uint4*>(orow + m * 64 + c * 8) = o;
    }
  }
}

// ---------------------------------------------------------------------------
// Host-side orchestration
// ---------------------------------------------------------------------------
extern "C" void kernel_launch(void* const* d_in, const int* in_sizes, int n_in,
                              void* d_out, int out_size, void* d_ws, size_t ws_size,
                              hipStream_t stream) {
  // setup_inputs dict order: x_v, Wqkv_v, bqkv_v, x_a, Wqkv_a, bqkv_a,
  //                          x_t, Wqkv_t, bqkv_t, Wo, bo
  const float* x[3]    = {(const float*)d_in[0], (const float*)d_in[3], (const float*)d_in[6]};
  const float* Wqkv[3] = {(const float*)d_in[1], (const float*)d_in[4], (const float*)d_in[7]};
  const float* bqkv[3] = {(const float*)d_in[2], (const float*)d_in[5], (const float*)d_in[8]};
  const float* Wo = (const float*)d_in[9];
  const float* bo = (const float*)d_in[10];
  float* out = (float*)d_out;

  const int R  = 8 * 2048;   // token rows = 16384
  const int IN = 512;
  const int ND = 3 * 1024;   // 3072

  // workspace carve-up (bf16 elements); total ~459 MiB
  unsigned short* p = (unsigned short*)d_ws;
  unsigned short* xb[3];   for (int m = 0; m < 3; ++m) { xb[m]   = p; p += (size_t)R * IN; }
  unsigned short* wqb[3];  for (int m = 0; m < 3; ++m) { wqb[m]  = p; p += (size_t)ND * IN; }
  unsigned short* wob = p; p += (size_t)ND * ND;
  unsigned short* qkvb[3]; for (int m = 0; m < 3; ++m) { qkvb[m] = p; p += (size_t)R * ND; }
  unsigned short* valsb = p;

  // 1) one-time f32 -> bf16 conversion of all GEMM operands
  for (int m = 0; m < 3; ++m) {
    int n4 = (R * IN) / 4;
    cvt_f32_to_bf16<<<(n4 + 255) / 256, 256, 0, stream>>>(x[m], xb[m], n4);
    n4 = (ND * IN) / 4;
    cvt_f32_to_bf16<<<(n4 + 255) / 256, 256, 0, stream>>>(Wqkv[m], wqb[m], n4);
  }
  {
    int n4 = (ND * ND) / 4;
    cvt_f32_to_bf16<<<(n4 + 255) / 256, 256, 0, stream>>>(Wo, wob, n4);
  }

  // 2) QKV GEMMs (bf16 out): qkv_m = x_m @ Wqkv_m^T + bqkv_m
  dim3 gemmBlk(256);
  dim3 gemmGrd(ND / BN, R / BM);
  for (int m = 0; m < 3; ++m)
    gemm_bf16_wmma<unsigned short><<<gemmGrd, gemmBlk, 0, stream>>>(
        xb[m], wqb[m], bqkv[m], qkvb[m], ND, IN);

  // 3) 3x3 cross-modal attention + residual (bf16 out)
  attn3x3_kernel<<<(R * 16) / 256, 256, 0, stream>>>(qkvb[0], qkvb[1], qkvb[2], valsb);

  // 4) output projection (f32 out): out = vals @ Wo^T + bo
  gemm_bf16_wmma<float><<<gemmGrd, gemmBlk, 0, stream>>>(
      valsb, wob, bo, out, ND, ND);
}
